// EpiGNN_41996190220980
// MI455X (gfx1250) — matrix operand order repaired and checked by reading, not compiled, not run on hardware
//
#include <hip/hip_runtime.h>

#define B_   8
#define T_   32
#define M_   1024
#define FIN  4
#define KCH  8
#define HIDR 40
#define HIDA 32
#define NHEAD 4
#define FH   10
#define TOUT 12

typedef __attribute__((ext_vector_type(16))) _Float16 v16h;
typedef __attribute__((ext_vector_type(8)))  float    v8f;

struct ConvParams {
    const float* w[5];
    const float* b[5];
    const float* g[5];
    const float* be[5];
};

__device__ __forceinline__ float warp_sum32(float v) {
#pragma unroll
    for (int off = 16; off >= 1; off >>= 1) v += __shfl_xor(v, off, 32);
    return v;
}
__device__ __forceinline__ float warp_max32(float v) {
#pragma unroll
    for (int off = 16; off >= 1; off >>= 1) v = fmaxf(v, __shfl_xor(v, off, 32));
    return v;
}

// ---------------------------------------------------------------- zero scratch
__global__ void k_zero(float* __restrict__ p, int n) {
    int i = blockIdx.x * 256 + threadIdx.x;
    if (i < n) p[i] = 0.f;
}

// ---------------------------------------------------------------- conv branches
// temp_emb[b,n,c] = tanh(tanh(max_t BN(conv)))   c = branch*8 + out_ch
__global__ void k_conv(const float* __restrict__ X, ConvParams cp,
                       float* __restrict__ temp) {
    __shared__ float xs[T_ * 64 * FIN];    // 32 KB  [t][n_local][f]
    __shared__ float swt[1536];
    __shared__ float ssc[40], ssh[40];

    const int bidx = blockIdx.x;           // 128 blocks
    const int b  = bidx >> 4;
    const int n0 = (bidx & 15) << 6;
    const int tid = threadIdx.x;           // 64 threads

    const float* xb = X + (size_t)b * T_ * M_ * FIN;
    for (int i = tid; i < T_ * 64 * FIN; i += 64) {
        int t = i >> 8; int rem = i & 255;
        xs[i] = xb[(size_t)t * (M_ * FIN) + n0 * FIN + rem];
    }
    const int wsz[5]  = {96, 160, 96, 160, 1024};
    const int woff[5] = {0, 96, 256, 352, 512};
    for (int br = 0; br < 5; ++br)
        for (int i = tid; i < wsz[br]; i += 64) swt[woff[br] + i] = cp.w[br][i];
    if (tid < 40) {
        int br = tid >> 3, c = tid & 7;
        float inv = rsqrtf(1.0f + 1e-5f);
        float sc = inv * cp.g[br][c];
        ssc[tid] = sc;
        ssh[tid] = cp.b[br][c] * sc + cp.be[br][c];
    }
    __syncthreads();

    const int nl = tid;
    const int ksA[5]  = {3, 5, 3, 5, 32};
    const int dilA[5] = {1, 1, 2, 2, 1};
    float* orow = temp + ((size_t)b * M_ + (n0 + nl)) * HIDR;
    for (int br = 0; br < 5; ++br) {
        const int ks = ksA[br], dil = dilA[br];
        const int Lout = T_ - dil * (ks - 1);
        for (int c = 0; c < KCH; ++c) {
            const float* wc = &swt[woff[br] + c * FIN * ks];
            float best = -3.4e38f;
            for (int t = 0; t < Lout; ++t) {
                float v = 0.f;
                for (int i = 0; i < FIN; ++i)
                    for (int kt = 0; kt < ks; ++kt)
                        v += xs[(t + kt * dil) * 256 + nl * 4 + i] * wc[i * ks + kt];
                best = fmaxf(best, v * ssc[br * 8 + c] + ssh[br * 8 + c]);
            }
            orow[br * 8 + c] = tanhf(tanhf(best));
        }
    }
}

// ---------------------------------------------------------------- q, K Gram, ksum
__global__ void k_qk(const float* __restrict__ temp,
                     const float* __restrict__ WQ, const float* __restrict__ bQ,
                     const float* __restrict__ WK, const float* __restrict__ bK,
                     float* __restrict__ qout, float* __restrict__ G,
                     float* __restrict__ ksum) {
    __shared__ float sWQ[HIDR * HIDA], sWK[HIDR * HIDA], sbQ[HIDA], sbK[HIDA];
    __shared__ float sk[256 * 33];
    const int tid = threadIdx.x;
    const int b = blockIdx.x >> 2;
    const int n = ((blockIdx.x & 3) << 8) + tid;

    for (int i = tid; i < HIDR * HIDA; i += 256) { sWQ[i] = WQ[i]; sWK[i] = WK[i]; }
    if (tid < HIDA) { sbQ[tid] = bQ[tid]; sbK[tid] = bK[tid]; }
    __syncthreads();

    float h[HIDR];
    const float* tr = temp + ((size_t)b * M_ + n) * HIDR;
#pragma unroll
    for (int i = 0; i < HIDR; ++i) h[i] = tr[i];
    float* qr = qout + ((size_t)b * M_ + n) * HIDA;
    for (int c = 0; c < HIDA; ++c) {
        float qv = sbQ[c], kv = sbK[c];
#pragma unroll
        for (int k = 0; k < HIDR; ++k) {
            qv += h[k] * sWQ[k * HIDA + c];
            kv += h[k] * sWK[k * HIDA + c];
        }
        qr[c] = qv;
        sk[tid * 33 + c] = kv;
    }
    __syncthreads();
    for (int e = 0; e < 4; ++e) {
        int idx = tid + (e << 8);
        int p = idx >> 5, qq = idx & 31;
        float s = 0.f;
        for (int r = 0; r < 256; ++r) s += sk[r * 33 + p] * sk[r * 33 + qq];
        atomicAdd(&G[b * 1024 + idx], s);
    }
    if (tid < 32) {
        float s = 0.f;
        for (int r = 0; r < 256; ++r) s += sk[r * 33 + tid];
        atomicAdd(&ksum[b * 32 + tid], s);
    }
}

// ---------------------------------------------------------------- adj bitmask + deg + feat_emb
__global__ void k_maskfeat(const float* __restrict__ adj, const float* __restrict__ temp,
                           const float* __restrict__ qin, const float* __restrict__ G,
                           const float* __restrict__ ksum,
                           const float* __restrict__ Wt, const float* __restrict__ bt,
                           const float* __restrict__ Ws, const float* __restrict__ bs,
                           float* __restrict__ feat, unsigned* __restrict__ mask) {
    __shared__ float sG[1024], sks[32], sWt[40], sbt[40], sWs[40], sbs[40];
    const int tid = threadIdx.x, lane = tid & 31, wave = tid >> 5;
    const int b = (blockIdx.x * 8) >> 10;
    const int n = (blockIdx.x * 8 + wave) & 1023;

    for (int i = tid; i < 1024; i += 256) sG[i] = G[b * 1024 + i];
    if (tid < 32) sks[tid] = ksum[b * 32 + tid];
    if (tid < 40) { sWt[tid] = Wt[tid]; sbt[tid] = bt[tid]; sWs[tid] = Ws[tid]; sbs[tid] = bs[tid]; }
    __syncthreads();

    const float* arow = adj + ((size_t)b * M_ + n) * M_;
    unsigned myword = 0u;
    float dsum = 0.f;
    for (int it = 0; it < 32; ++it) {
        float av = arow[it * 32 + lane];
        dsum += av;
        unsigned bal = (unsigned)__ballot(av > 0.f);
        if (lane == it) myword = bal;
    }
    mask[((size_t)b * M_ + n) * 32 + lane] = myword;
    dsum = warp_sum32(dsum);

    float qv  = qin[((size_t)b * M_ + n) * HIDA + lane];
    float qdk = warp_sum32(qv * sks[lane]);
    float gq = 0.f;
#pragma unroll
    for (int qq = 0; qq < 32; ++qq) gq += sG[lane * 32 + qq] * __shfl(qv, qq, 32);
    float qGq = warp_sum32(qv * gq);
    float t_in = qdk / fmaxf(sqrtf(fmaxf(qGq, 0.f)), 1e-12f);

    const float* trow = temp + ((size_t)b * M_ + n) * HIDR;
    float* frow = feat + ((size_t)b * M_ + n) * HIDR;
    for (int c = lane; c < HIDR; c += 32)
        frow[c] = trow[c] + t_in * sWt[c] + sbt[c] + dsum * sWs[c] + sbs[c];
}

// ---------------------------------------------------------------- GAT: Wh (f16), el, er
__global__ void k_gatpre(const float* __restrict__ state, const float* __restrict__ W,
                         const float* __restrict__ a, _Float16* __restrict__ whf,
                         float* __restrict__ el, float* __restrict__ er) {
    __shared__ float sW[HIDR * HIDR], sa[NHEAD * 2 * FH];
    const int tid = threadIdx.x;
    const int idx = blockIdx.x * 256 + tid;
    const int b = idx >> 10, n = idx & 1023;
    for (int i = tid; i < HIDR * HIDR; i += 256) sW[i] = W[i];
    if (tid < NHEAD * 2 * FH) sa[tid] = a[tid];
    __syncthreads();

    float h[HIDR];
    const float* sr = state + (size_t)idx * HIDR;
#pragma unroll
    for (int i = 0; i < HIDR; ++i) h[i] = sr[i];
    float wh[HIDR];
    for (int c = 0; c < HIDR; ++c) {
        float s = 0.f;
#pragma unroll
        for (int k = 0; k < HIDR; ++k) s += h[k] * sW[k * HIDR + c];
        wh[c] = s;
    }
    for (int hd = 0; hd < NHEAD; ++hd) {
        float e1 = 0.f, e2 = 0.f;
#pragma unroll
        for (int f = 0; f < FH; ++f) {
            e1 += wh[hd * FH + f] * sa[hd * 2 * FH + f];
            e2 += wh[hd * FH + f] * sa[hd * 2 * FH + FH + f];
        }
        size_t o = (size_t)(b * NHEAD + hd) * M_ + n;
        el[o] = e1; er[o] = e2;
        _Float16* wr = whf + o * 16;
#pragma unroll
        for (int f = 0; f < 16; ++f) wr[f] = (_Float16)(f < FH ? wh[hd * FH + f] : 0.f);
    }
}

// ---------------------------------------------------------------- GAT attention (WMMA f16)
// block = (b, h, 128-row tile), 8 waves; each wave owns 16 rows, K loop over 1024 cols.
// Wh is staged into LDS pre-swizzled to the exact 16x16x32 B-fragment layout so each
// lane reads its whole fragment as one contiguous 32B v16h load.
__global__ void __launch_bounds__(256) k_gatattn(const _Float16* __restrict__ whf,
                                                 const float* __restrict__ el,
                                                 const float* __restrict__ er,
                                                 const unsigned* __restrict__ mask,
                                                 float* __restrict__ gout) {
    __shared__ __align__(16) _Float16 sWhF[M_ * 16];   // 32 KB, fragment order
    __shared__ float sEr[M_];                          // 4 KB
    __shared__ float sEl[128];
    __shared__ unsigned sMask[128 * 32];               // 16 KB
    __shared__ float sRowmax[128];
    __shared__ float sDenom[128];

    const int tid = threadIdx.x, lane = tid & 31, wave = tid >> 5;
    const int blk = blockIdx.x;                        // 256 blocks
    const int b  = blk >> 5;
    const int h  = (blk >> 3) & 3;
    const int n0 = (blk & 7) << 7;
    const size_t bh = (size_t)(b * NHEAD + h) * M_;

    // stage Wh rows -> B-fragment-order LDS:
    //   element (m, f) -> sWhF[(m>>4)*256 + f*16 + (m&15)]
    for (int m = tid; m < M_; m += 256) {
        const uint4* src = (const uint4*)(whf + (bh + m) * 16);
        _Float16 vals[16];
        *(uint4*)&vals[0] = src[0];
        *(uint4*)&vals[8] = src[1];
        const int base = (m >> 4) * 256 + (m & 15);
#pragma unroll
        for (int f = 0; f < 16; ++f)
            sWhF[base + f * 16] = vals[f];
    }
    for (int i = tid; i < M_; i += 256) sEr[i] = er[bh + i];
    if (tid < 128) sEl[tid] = el[bh + n0 + tid];
    for (int i = tid; i < 128 * 32; i += 256) {
        int r = i >> 5, w = i & 31;
        sMask[i] = mask[((size_t)b * M_ + n0 + r) * 32 + w];
    }
    __syncthreads();

    // per-row softmax max = lrelu(el + max_{masked} er), or -1e30 if row empty
    for (int rr = 0; rr < 16; ++rr) {
        const int row = wave * 16 + rr;
        const unsigned* mrow = &sMask[row * 32];
        float mx = -3.4e38f;
        for (int m = lane; m < M_; m += 32)
            if ((mrow[m >> 5] >> (m & 31)) & 1u) mx = fmaxf(mx, sEr[m]);
        mx = warp_max32(mx);
        if (lane == 0) {
            float e;
            if (mx < -1e37f) e = -1e30f;
            else { e = sEl[row] + mx; e = (e >= 0.f) ? e : 0.2f * e; }
            sRowmax[row] = e;
        }
    }
    __syncthreads();

    const int Mr = lane & 15, hi = lane >> 4;
    const int lrow = wave * 16 + Mr;
    const float myEl = sEl[lrow];
    const float myMax = sRowmax[lrow];
    const unsigned* myMask = &sMask[lrow * 32];

    v8f acc = {};
    float psum = 0.f;
    for (int kt = 0; kt < 32; ++kt) {
        const int kb = kt << 5;
        const unsigned mw = myMask[kt];     // tile's 32 mask bits (mm>>5 == kt)
        v16h afrag;
        // A fragment: lane holds row Mr; K slots contiguous, K = 16*hi + s (matches B)
#pragma unroll
        for (int v = 0; v < 8; ++v) {
#pragma unroll
            for (int j = 0; j < 2; ++j) {
                const int K = (2 * v + j) + (hi << 4);
                float ee = myEl + sEr[kb + K];
                ee = (ee >= 0.f) ? ee : 0.2f * ee;
                float e = ((mw >> K) & 1u) ? ee : -1e30f;
                float p = __expf(e - myMax);
                psum += p;
                afrag[2 * v + j] = (_Float16)p;
            }
        }
        // B fragment: contiguous 32B per lane in fragment-order LDS
        const v16h bfrag = *(const v16h*)(sWhF + (size_t)(kb + lane) * 16);
        acc = __builtin_amdgcn_wmma_f32_16x16x32_f16(false, afrag, false, bfrag,
                                                     (short)0, acc, false, false);
    }
    psum += __shfl_xor(psum, 16, 32);
    if (hi == 0) sDenom[lrow] = psum;
    __syncthreads();

    const int f = Mr;
#pragma unroll
    for (int r = 0; r < 8; ++r) {
        const int row = wave * 16 + r + (hi << 3);
        // fast reciprocal (v_rcp_f32): softmax normalization, ~1 ulp is plenty
        float vv = acc[r] * __builtin_amdgcn_rcpf(sDenom[row]);
        vv = (vv > 0.f) ? vv : (__expf(vv) - 1.0f);   // ELU
        if (f < FH)
            gout[((size_t)b * M_ + n0 + row) * HIDR + h * FH + f] = vv;
    }
}

// ---------------------------------------------------------------- output projection (+transpose)
__global__ void k_out(const float* __restrict__ g1, const float* __restrict__ g2,
                      const float* __restrict__ g3, const float* __restrict__ feat,
                      const float* __restrict__ Wout, const float* __restrict__ bout,
                      float* __restrict__ out) {
    __shared__ float sW[4 * HIDR * TOUT], sb[TOUT];
    const int tid = threadIdx.x;
    const int idx = blockIdx.x * 256 + tid;
    const int b = idx >> 10, n = idx & 1023;
    for (int i = tid; i < 4 * HIDR * TOUT; i += 256) sW[i] = Wout[i];
    if (tid < TOUT) sb[tid] = bout[tid];
    __syncthreads();

    float acc[TOUT];
#pragma unroll
    for (int t = 0; t < TOUT; ++t) acc[t] = sb[t];
    const float* parts[4] = { g1 + (size_t)idx * HIDR, g2 + (size_t)idx * HIDR,
                              g3 + (size_t)idx * HIDR, feat + (size_t)idx * HIDR };
    for (int p = 0; p < 4; ++p) {
        const float* hr = parts[p];
        for (int k = 0; k < HIDR; ++k) {
            float hv = hr[k];
            const float* wr = &sW[(p * HIDR + k) * TOUT];
#pragma unroll
            for (int t = 0; t < TOUT; ++t) acc[t] += hv * wr[t];
        }
    }
#pragma unroll
    for (int t = 0; t < TOUT; ++t)
        out[((size_t)b * TOUT + t) * M_ + n] = acc[t];
}

// ---------------------------------------------------------------- host
extern "C" void kernel_launch(void* const* d_in, const int* in_sizes, int n_in,
                              void* d_out, int out_size, void* d_ws, size_t ws_size,
                              hipStream_t stream) {
    const float* X   = (const float*)d_in[0];
    const float* adj = (const float*)d_in[1];
    ConvParams cp;
    for (int i = 0; i < 5; ++i) {
        cp.w[i]  = (const float*)d_in[2 + 4 * i];
        cp.b[i]  = (const float*)d_in[3 + 4 * i];
        cp.g[i]  = (const float*)d_in[4 + 4 * i];
        cp.be[i] = (const float*)d_in[5 + 4 * i];
    }
    const float* WQ = (const float*)d_in[22]; const float* bQ = (const float*)d_in[23];
    const float* WK = (const float*)d_in[24]; const float* bK = (const float*)d_in[25];
    const float* Wt = (const float*)d_in[26]; const float* bt = (const float*)d_in[27];
    const float* Ws = (const float*)d_in[28]; const float* bs = (const float*)d_in[29];
    // d_in[30..33] (Wg1,bg1,Wg2,bg2) are dead code in the reference.
    const float* gatW[3] = { (const float*)d_in[34], (const float*)d_in[36], (const float*)d_in[38] };
    const float* gatA[3] = { (const float*)d_in[35], (const float*)d_in[37], (const float*)d_in[39] };
    const float* Wout = (const float*)d_in[40];
    const float* bout = (const float*)d_in[41];
    float* out = (float*)d_out;

    float* ws = (float*)d_ws;
    const size_t o_temp = 0;                       // 8*1024*40
    const size_t o_q    = o_temp + 327680;         // 8*1024*32
    const size_t o_G    = o_q    + 262144;         // 8*1024
    const size_t o_ks   = o_G    + 8192;           // 8*32
    const size_t o_feat = o_ks   + 256;            // 8*1024*40
    const size_t o_mask = o_feat + 327680;         // 8*1024*32 u32
    const size_t o_el   = o_mask + 262144;         // 8*4*1024
    const size_t o_er   = o_el   + 32768;          // 8*4*1024
    const size_t o_g1   = o_er   + 32768;          // 3 x 8*1024*40
    const size_t o_g2   = o_g1   + 327680;
    const size_t o_g3   = o_g2   + 327680;
    const size_t o_whf  = o_g3   + 327680;         // 8*4*1024*16 halfs (262144 float slots)

    float* temp = ws + o_temp;
    float* q    = ws + o_q;
    float* G    = ws + o_G;
    float* ksum = ws + o_ks;
    float* feat = ws + o_feat;
    unsigned* mask = (unsigned*)(ws + o_mask);
    float* el   = ws + o_el;
    float* er   = ws + o_er;
    float* gl[3] = { ws + o_g1, ws + o_g2, ws + o_g3 };
    _Float16* whf = (_Float16*)(ws + o_whf);

    k_zero<<<33, 256, 0, stream>>>(G, 8192 + 256);                 // G + ksum contiguous
    k_conv<<<128, 64, 0, stream>>>(X, cp, temp);
    k_qk<<<32, 256, 0, stream>>>(temp, WQ, bQ, WK, bK, q, G, ksum);
    k_maskfeat<<<1024, 256, 0, stream>>>(adj, temp, q, G, ksum, Wt, bt, Ws, bs, feat, mask);

    const float* state = feat;
    for (int L = 0; L < 3; ++L) {
        k_gatpre<<<32, 256, 0, stream>>>(state, gatW[L], gatA[L], whf, el, er);
        k_gatattn<<<256, 256, 0, stream>>>(whf, el, er, mask, gl[L]);
        state = gl[L];
    }
    k_out<<<32, 256, 0, stream>>>(gl[0], gl[1], gl[2], feat, Wout, bout, out);
}
